// GraphAttentionNetwork_51797305590027
// MI455X (gfx1250) — matrix operand order repaired and checked by reading, not compiled, not run on hardware
//
#include <hip/hip_runtime.h>

// ---------------------------------------------------------------------------
// GAT network for gfx1250 (MI455X): WMMA f16 GEMMs + streaming edge passes.
// ---------------------------------------------------------------------------

typedef __attribute__((ext_vector_type(16))) _Float16 v16h;
typedef __attribute__((ext_vector_type(8)))  _Float16 v8h;
typedef __attribute__((ext_vector_type(8)))  float    v8f;

#define HID 64
#define NH  4
#define CH  256   // NH*HID

// ---- helpers ---------------------------------------------------------------

__device__ __forceinline__ unsigned f2ord(float f) {
  unsigned u = __float_as_uint(f);
  return (u & 0x80000000u) ? ~u : (u | 0x80000000u);
}
__device__ __forceinline__ float ord2f(unsigned u) {
  return __uint_as_float((u & 0x80000000u) ? (u & 0x7fffffffu) : ~u);
}
__device__ __forceinline__ float lrelu(float x) { return x > 0.f ? x : 0.2f * x; }
__device__ __forceinline__ float eluf(float x)  { return x > 0.f ? x : (__expf(x) - 1.f); }

// ---- fills -----------------------------------------------------------------

__global__ void k_fill_f32(float* p, float v, int n) {
  int t = blockIdx.x * blockDim.x + threadIdx.x;
  if (t < n) p[t] = v;
}
__global__ void k_fill_u32(unsigned* p, unsigned v, int n) {
  int t = blockIdx.x * blockDim.x + threadIdx.x;
  if (t < n) p[t] = v;
}

// ---- precision conversion / weight transpose -------------------------------

__global__ void k_f32_to_f16(const float* __restrict__ in, _Float16* __restrict__ out, int n) {
  int t = blockIdx.x * blockDim.x + threadIdx.x;
  if (t < n) out[t] = (_Float16)in[t];
}

// W[64,256] (row-major) -> WT[256,64] f16 so B fragments are contiguous per lane
__global__ void k_transpose_w(const float* __restrict__ W, _Float16* __restrict__ WT) {
  int t = blockIdx.x * blockDim.x + threadIdx.x;   // 64*256 = 16384
  if (t >= 64 * CH) return;
  int k = t >> 8, n = t & 255;
  WT[n * 64 + k] = (_Float16)W[k * CH + n];
}

// ---- layer-0 GEMM: hw[N,256] = x[N,9] @ W0[9,256] (tiny K, full f32) -------

__global__ void k_gemm0(const float* __restrict__ x, const float* __restrict__ W,
                        float* __restrict__ hw) {
  __shared__ float sx[9];
  int node = blockIdx.x;
  int j = threadIdx.x;                  // 0..255
  if (j < 9) sx[j] = x[node * 9 + j];
  __syncthreads();
  float acc = 0.f;
#pragma unroll
  for (int k = 0; k < 9; ++k) acc += sx[k] * W[k * CH + j];
  hw[node * CH + j] = acc;
}

// ---- main GEMM: hw[N,256] = hh[N,64](f16) @ W[64,256] via WMMA -------------
// Wave handles one 16-row tile x two 16-col tiles; 8 waves/block cover 256 cols.
// A fragment (16x32 f16): lane l, m = l&15, half=l>>4:
//   a[i] = A[m][kk + 16*half + i]  (i = 0..15)
// B fragment (32x16 f16) from WT[256,64]: lane col n = l&15, kb = kk + 16*half:
//   b[i] = WT[n][kb+i]  (32 contiguous bytes)
__global__ void k_gemm_wmma(const _Float16* __restrict__ A, const _Float16* __restrict__ WT,
                            float* __restrict__ C) {
  int lane = threadIdx.x & 31;
  int wave = threadIdx.x >> 5;
  int rbase = blockIdx.x * 16;
  int m = rbase + (lane & 15);
  int half = lane >> 4;
  int n0 = wave * 32 + (lane & 15);     // col tile 2*wave
  int n1 = n0 + 16;                     // col tile 2*wave+1

  v8f acc0 = {}; v8f acc1 = {};
#pragma unroll
  for (int kk = 0; kk < HID; kk += 32) {
    const _Float16* arow = A + (size_t)m * HID + kk + half * 16;
    v8h lo = *(const v8h*)arow;
    v8h hi = *(const v8h*)(arow + 8);
    v16h a;
#pragma unroll
    for (int i = 0; i < 8; ++i) { a[i] = lo[i]; a[i + 8] = hi[i]; }

    int kb = kk + half * 16;
    v16h b0 = *(const v16h*)(WT + n0 * HID + kb);
    v16h b1 = *(const v16h*)(WT + n1 * HID + kb);

    acc0 = __builtin_amdgcn_wmma_f32_16x16x32_f16(false, a, false, b0, (short)0, acc0, false, false);
    acc1 = __builtin_amdgcn_wmma_f32_16x16x32_f16(false, a, false, b1, (short)0, acc1, false, false);
  }
  int row_off = half * 8;
#pragma unroll
  for (int v = 0; v < 8; ++v) {
    int r = rbase + row_off + v;
    C[(size_t)r * CH + n0] = acc0[v];
    C[(size_t)r * CH + n1] = acc1[v];
  }
}

// ---- attention logits: es/ed[N,4] (wave per (node,head)) -------------------

__global__ void k_scores(const float* __restrict__ hw, const float* __restrict__ asrc,
                         const float* __restrict__ adst, float* __restrict__ es,
                         float* __restrict__ ed, int n) {
  int lane = threadIdx.x & 31;
  int wid = blockIdx.x * (blockDim.x >> 5) + (threadIdx.x >> 5);
  if (wid >= n * NH) return;
  int node = wid >> 2, hd = wid & 3;
  const float* row = hw + (size_t)node * CH + hd * HID;
  const float* as = asrc + hd * HID;
  const float* ad = adst + hd * HID;
  float r0 = row[lane], r1 = row[lane + 32];
  float s = r0 * as[lane] + r1 * as[lane + 32];
  float d = r0 * ad[lane] + r1 * ad[lane + 32];
#pragma unroll
  for (int off = 16; off > 0; off >>= 1) {
    s += __shfl_xor(s, off);
    d += __shfl_xor(d, off);
  }
  if (lane == 0) { es[wid] = s; ed[wid] = d; }
}

// ---- edge passes (self-loops: e in [E, E+n) => s=d=e-E) --------------------

__global__ void k_edge_max(const int* __restrict__ ei, const float* __restrict__ es,
                           const float* __restrict__ ed, unsigned* __restrict__ amax,
                           int E, int n) {
  int e = blockIdx.x * blockDim.x + threadIdx.x;
  if (e >= E + n) return;
  int s = (e < E) ? ei[e] : (e - E);
  int d = (e < E) ? ei[E + e] : (e - E);
#pragma unroll
  for (int hd = 0; hd < NH; ++hd) {
    float a = lrelu(es[s * NH + hd] + ed[d * NH + hd]);
    atomicMax(&amax[d * NH + hd], f2ord(a));
  }
}

__global__ void k_edge_sum(const int* __restrict__ ei, const float* __restrict__ es,
                           const float* __restrict__ ed, const unsigned* __restrict__ amax,
                           float* __restrict__ denom, int E, int n) {
  int e = blockIdx.x * blockDim.x + threadIdx.x;
  if (e >= E + n) return;
  int s = (e < E) ? ei[e] : (e - E);
  int d = (e < E) ? ei[E + e] : (e - E);
#pragma unroll
  for (int hd = 0; hd < NH; ++hd) {
    float a = lrelu(es[s * NH + hd] + ed[d * NH + hd]);
    atomicAdd(&denom[d * NH + hd], __expf(a - ord2f(amax[d * NH + hd])));
  }
}

__global__ void k_edge_w(const int* __restrict__ ei, const float* __restrict__ es,
                         const float* __restrict__ ed, const unsigned* __restrict__ amax,
                         const float* __restrict__ denom, float* __restrict__ wbuf,
                         int E, int n) {
  int e = blockIdx.x * blockDim.x + threadIdx.x;
  if (e >= E + n) return;
  int s = (e < E) ? ei[e] : (e - E);
  int d = (e < E) ? ei[E + e] : (e - E);
#pragma unroll
  for (int hd = 0; hd < NH; ++hd) {
    float a = lrelu(es[s * NH + hd] + ed[d * NH + hd]);
    wbuf[e * NH + hd] = __expf(a - ord2f(amax[d * NH + hd])) / denom[d * NH + hd];
  }
}

// ---- message scatter: thread per (edge, channel) ---------------------------

__global__ void k_scatter(const int* __restrict__ ei, const float* __restrict__ hw,
                          const float* __restrict__ wbuf, float* __restrict__ agg,
                          int E, int n) {
  int t = blockIdx.x * blockDim.x + threadIdx.x;
  if (t >= (E + n) * HID) return;
  int e = t >> 6, c = t & 63;
  int s = (e < E) ? ei[e] : (e - E);
  int d = (e < E) ? ei[E + e] : (e - E);
#pragma unroll
  for (int hd = 0; hd < NH; ++hd) {
    float msg = hw[(size_t)s * CH + hd * HID + c] * wbuf[e * NH + hd];
    atomicAdd(&agg[(size_t)d * CH + hd * HID + c], msg);
  }
}

// ---- head mean + bias + BN + ELU + residual (in place on h) ----------------

__global__ void k_finalize(const float* __restrict__ agg, const float* __restrict__ bias,
                           const float* __restrict__ gamma, const float* __restrict__ beta,
                           const float* __restrict__ mean, const float* __restrict__ var,
                           float* __restrict__ h, int residual, int n) {
  int t = blockIdx.x * blockDim.x + threadIdx.x;
  if (t >= n * HID) return;
  int node = t >> 6, c = t & 63;
  const float* a = agg + (size_t)node * CH + c;
  float v = 0.25f * (a[0] + a[64] + a[128] + a[192]) + bias[c];
  v = (v - mean[c]) * rsqrtf(var[c] + 1e-5f) * gamma[c] + beta[c];
  v = eluf(v);
  h[t] = residual ? (h[t] + v) : v;
}

// ---- global mean pool ------------------------------------------------------

__global__ void k_pool(const float* __restrict__ h, const long long* __restrict__ batch,
                       float* __restrict__ pool, float* __restrict__ cnt, int n) {
  int t = blockIdx.x * blockDim.x + threadIdx.x;
  if (t >= n * HID) return;
  int node = t >> 6, c = t & 63;
  int g = (int)batch[node];
  atomicAdd(&pool[(size_t)g * HID + c], h[t]);
  if (c == 0) atomicAdd(&cnt[g], 1.f);
}

// ---- output MLP ------------------------------------------------------------

__global__ void k_mlp(const float* __restrict__ pool, const float* __restrict__ cnt,
                      const float* __restrict__ W1, const float* __restrict__ b1,
                      const float* __restrict__ W2, const float* __restrict__ b2,
                      float* __restrict__ out, int G) {
  int g = blockIdx.x * blockDim.x + threadIdx.x;
  if (g >= G) return;
  float inv = 1.f / fmaxf(cnt[g], 1.f);
  float p[HID];
#pragma unroll
  for (int c = 0; c < HID; ++c) p[c] = pool[(size_t)g * HID + c] * inv;
  float acc = b2[0];
#pragma unroll 4
  for (int j = 0; j < 32; ++j) {
    float z = b1[j];
#pragma unroll
    for (int c = 0; c < HID; ++c) z += p[c] * W1[c * 32 + j];
    acc += eluf(z) * W2[j];
  }
  out[g] = acc;
}

// ---------------------------------------------------------------------------

extern "C" void kernel_launch(void* const* d_in, const int* in_sizes, int n_in,
                              void* d_out, int out_size, void* d_ws, size_t ws_size,
                              hipStream_t stream) {
  const float*     x     = (const float*)d_in[0];
  const int*       ei    = (const int*)d_in[1];
  const long long* batch = (const long long*)d_in[2];
  const float* g0W  = (const float*)d_in[3];
  const float* g0as = (const float*)d_in[4];
  const float* g0ad = (const float*)d_in[5];
  const float* g0b  = (const float*)d_in[6];
  const float* bn0g = (const float*)d_in[7];
  const float* bn0b = (const float*)d_in[8];
  const float* bn0m = (const float*)d_in[9];
  const float* bn0v = (const float*)d_in[10];
  const float* gW   = (const float*)d_in[11];
  const float* gas  = (const float*)d_in[12];
  const float* gad  = (const float*)d_in[13];
  const float* gb   = (const float*)d_in[14];
  const float* bng  = (const float*)d_in[15];
  const float* bnb  = (const float*)d_in[16];
  const float* bnm  = (const float*)d_in[17];
  const float* bnv  = (const float*)d_in[18];
  const float* W1   = (const float*)d_in[19];
  const float* b1   = (const float*)d_in[20];
  const float* W2   = (const float*)d_in[21];
  const float* b2   = (const float*)d_in[22];

  const int N  = in_sizes[0] / 9;
  const int E  = in_sizes[1] / 2;
  const int G  = out_size;
  const int EP = E + N;

  // workspace carve-out
  char* ws = (char*)d_ws;
  size_t off = 0;
  auto alloc = [&](size_t bytes) -> void* {
    void* p = ws + off;
    off = (off + bytes + 255) & ~(size_t)255;
    return p;
  };
  float*     hw    = (float*)alloc((size_t)N * CH * 4);
  float*     agg   = (float*)alloc((size_t)N * CH * 4);
  float*     h     = (float*)alloc((size_t)N * HID * 4);
  _Float16*  hh    = (_Float16*)alloc((size_t)N * HID * 2);
  float*     es    = (float*)alloc((size_t)N * NH * 4);
  float*     ed    = (float*)alloc((size_t)N * NH * 4);
  unsigned*  amax  = (unsigned*)alloc((size_t)N * NH * 4);
  float*     denom = (float*)alloc((size_t)N * NH * 4);
  float*     wbuf  = (float*)alloc((size_t)EP * NH * 4);
  _Float16*  WT    = (_Float16*)alloc((size_t)CH * HID * 2);
  float*     pool  = (float*)alloc((size_t)G * HID * 4);
  float*     cnt   = (float*)alloc((size_t)G * 4);
  (void)ws_size;

  const int B = 256;
  auto cdiv = [](int a, int b) { return (a + b - 1) / b; };

  // runs the attention + aggregation machinery for the current hw[]
  auto run_attention = [&](const float* asrc, const float* adst, const float* bias,
                           const float* bg, const float* bb, const float* bm,
                           const float* bv, int residual) {
    k_scores<<<cdiv(N * NH, B / 32), B, 0, stream>>>(hw, asrc, adst, es, ed, N);
    k_fill_u32<<<cdiv(N * NH, B), B, 0, stream>>>(amax, 0u, N * NH);
    k_fill_f32<<<cdiv(N * NH, B), B, 0, stream>>>(denom, 0.f, N * NH);
    k_fill_f32<<<cdiv(N * CH, B), B, 0, stream>>>(agg, 0.f, N * CH);
    k_edge_max<<<cdiv(EP, B), B, 0, stream>>>(ei, es, ed, amax, E, N);
    k_edge_sum<<<cdiv(EP, B), B, 0, stream>>>(ei, es, ed, amax, denom, E, N);
    k_edge_w<<<cdiv(EP, B), B, 0, stream>>>(ei, es, ed, amax, denom, wbuf, E, N);
    k_scatter<<<cdiv(EP * HID, B), B, 0, stream>>>(ei, hw, wbuf, agg, E, N);
    k_finalize<<<cdiv(N * HID, B), B, 0, stream>>>(agg, bias, bg, bb, bm, bv, h, residual, N);
  };

  // ---- layer 0 (K=9, scalar GEMM) ----
  k_gemm0<<<N, B, 0, stream>>>(x, g0W, hw);
  run_attention(g0as, g0ad, g0b, bn0g, bn0b, bn0m, bn0v, /*residual=*/0);

  // ---- layers 1..3 (WMMA GEMM, residual) ----
  for (int i = 0; i < 3; ++i) {
    k_transpose_w<<<cdiv(64 * CH, B), B, 0, stream>>>(gW + (size_t)i * 64 * CH, WT);
    k_f32_to_f16<<<cdiv(N * HID, B), B, 0, stream>>>(h, hh, N * HID);
    k_gemm_wmma<<<cdiv(N, 16), B, 0, stream>>>(hh, WT, hw);
    run_attention(gas + i * NH * HID, gad + i * NH * HID, gb + i * HID,
                  bng + i * HID, bnb + i * HID, bnm + i * HID, bnv + i * HID,
                  /*residual=*/1);
  }

  // ---- global mean pool + MLP ----
  k_fill_f32<<<cdiv(G * HID, B), B, 0, stream>>>(pool, 0.f, G * HID);
  k_fill_f32<<<cdiv(G, B), B, 0, stream>>>(cnt, 0.f, G);
  k_pool<<<cdiv(N * HID, B), B, 0, stream>>>(h, batch, pool, cnt, N);
  k_mlp<<<cdiv(G, B), B, 0, stream>>>(pool, cnt, W1, b1, W2, b2, (float*)d_out, G);
}
